// BiLSTM_Hard_CRF_51144470561407
// MI455X (gfx1250) — matrix operand order repaired and checked by reading, not compiled, not run on hardware
//
#include <hip/hip_runtime.h>
#include <hip/hip_bf16.h>
#include <math.h>

typedef __attribute__((ext_vector_type(16))) _Float16 v16h;
typedef __attribute__((ext_vector_type(8)))  _Float16 v8h;
typedef __attribute__((ext_vector_type(8)))  float    v8f;

#define CRF_B 256
#define CRF_S 2048
#define CRF_T 64
#define ROWS  16            // batch rows per block
#define APAD  (CRF_T + 4)   // padded alpha row stride (floats)
#define FEAT_BYTES (ROWS * CRF_T * 4)   // one f_t tile: 4096 B
#define LN2F  0.6931471805599453f

// ---------------------------------------------------------------------------
// Kernel 1: gold path score per batch element -> d_ws (float[B])
// ---------------------------------------------------------------------------
__global__ void crf_gold_kernel(const float* __restrict__ feats,
                                const float* __restrict__ trans,
                                const float* __restrict__ startt,
                                const float* __restrict__ endt,
                                const int*   __restrict__ tags,
                                const unsigned char* __restrict__ mask,
                                float* __restrict__ gold) {
    const int b   = blockIdx.x;
    const int tid = threadIdx.x;

    __shared__ float s_trans[CRF_T * CRF_T];
    __shared__ float s_sum[256];
    __shared__ int   s_cnt[256];

    for (int i = tid; i < CRF_T * CRF_T; i += blockDim.x) s_trans[i] = trans[i];
    __syncthreads();

    const int*           tb = tags  + (size_t)b * CRF_S;
    const unsigned char* mb = mask  + (size_t)b * CRF_S;
    const float*         fb = feats + (size_t)b * CRF_S * CRF_T;

    float acc = 0.0f;
    int   cnt = 0;
    for (int s = tid; s < CRF_S; s += blockDim.x) {
        const int tg = tb[s];
        const bool m = (mb[s] != 0);
        cnt += m ? 1 : 0;
        if (s < CRF_S - 1) {
            if (m) acc += fb[(size_t)s * CRF_T + tg];                 // emit * mask[:, :-1]
            if (mb[s + 1]) acc += s_trans[tg * CRF_T + tb[s + 1]];    // trans * mask[:, 1:]
        }
    }
    s_sum[tid] = acc;
    s_cnt[tid] = cnt;
    __syncthreads();
    for (int off = 128; off > 0; off >>= 1) {
        if (tid < off) { s_sum[tid] += s_sum[tid + off]; s_cnt[tid] += s_cnt[tid + off]; }
        __syncthreads();
    }
    if (tid == 0) {
        const int last = s_cnt[0] - 1;
        const int lt   = tb[last];
        float g = startt[tb[0]] + s_sum[0] + endt[lt];
        if (mb[CRF_S - 1]) g += fb[(size_t)(CRF_S - 1) * CRF_T + lt];
        gold[b] = g;
    }
}

// ---------------------------------------------------------------------------
// Kernel 2: forward scan via WMMA exp-space GEMM, async-LDS double buffering.
// grid = B/16, block = 128 (4 waves; wave w owns states [16w, 16w+16)).
// ---------------------------------------------------------------------------
__global__ void __launch_bounds__(128)
crf_forward_kernel(const float* __restrict__ feats,
                   const float* __restrict__ trans,
                   const float* __restrict__ startt,
                   const float* __restrict__ endt,
                   const unsigned char* __restrict__ mask,
                   const float* __restrict__ gold,
                   float* __restrict__ out) {
    const int tid    = threadIdx.x;
    const int lane   = tid & 31;
    const int wave   = tid >> 5;        // 0..3 -> N tile
    const int laneHi = lane >> 4;       // 0/1: lane half
    const int nn     = lane & 15;
    const int Nbase  = wave * 16;
    const int bb     = blockIdx.x * ROWS;

    __shared__ float                  sh_alpha[ROWS][APAD];
    __shared__ __align__(16) _Float16 sh_expA[ROWS][CRF_T];
    __shared__ float                  sh_rowmax[ROWS];
    __shared__ float                  sh_pmax[ROWS][8];
    __shared__ float                  sh_mrowf[ROWS];
    __shared__ __align__(16) unsigned char sh_mask[ROWS * CRF_S];       // 32 KB
    __shared__ __align__(16) float    sh_feat[2][ROWS][CRF_T];          // 8 KB

    // ---- preload the block's mask slab (rows bb..bb+15 are contiguous) ----
    {
        const uint4* msrc = (const uint4*)(mask + (size_t)bb * CRF_S);
        uint4*       mdst = (uint4*)sh_mask;
        for (int i = tid; i < (ROWS * CRF_S) / 16; i += 128) mdst[i] = msrc[i];
    }

    // ---- resident B fragments: exp(trans) as f16, WMMA 32x16 layout -------
    v16h bf0, bf1;
    #pragma unroll
    for (int h = 0; h < 16; ++h) {
        const int K = (h & 7) + 8 * laneHi + 16 * (h >> 3);
        bf0[h] = (_Float16)__expf(trans[(K)      * CRF_T + Nbase + nn]);
        bf1[h] = (_Float16)__expf(trans[(K + 32) * CRF_T + Nbase + nn]);
    }

    // ---- alpha0 = start + feats[:,0,:] ------------------------------------
    #pragma unroll
    for (int e = 0; e < 8; ++e) {
        const int idx = tid * 8 + e;         // 0..1023
        const int rr  = idx >> 6;
        const int cc  = idx & 63;
        sh_alpha[rr][cc] = startt[cc] + feats[((size_t)(bb + rr) * CRF_S) * CRF_T + cc];
    }

    // ---- async double-buffer bootstrap: feats[:,1,:] -> buffer 1 ----------
    // Each thread copies 32 B (two b128): rows tid>>3, cols (tid&7)*8.
    const unsigned ldsFeatBase =
        (unsigned)(unsigned long long)(uintptr_t)&sh_feat[0][0][0];
    unsigned long long gaddr = (unsigned long long)(uintptr_t)
        (feats + ((size_t)(bb + (tid >> 3)) * CRF_S + 1) * CRF_T + (size_t)(tid & 7) * 8);
    {
        const unsigned dst = ldsFeatBase + FEAT_BYTES + tid * 32u;  // buf 1
        asm volatile(
            "global_load_async_to_lds_b128 %0, %1, off\n\t"
            "global_load_async_to_lds_b128 %0, %1, off offset:16"
            :: "v"(dst), "v"(gaddr) : "memory");
        gaddr += (unsigned long long)CRF_T * sizeof(float);
    }
    __syncthreads();

    const int Arow = lane & 15;              // A-fragment row

    for (int t = 1; t < CRF_S; ++t) {
        const int buf = t & 1;

        // -- kick off async copy of feats[:, t+1, :] into the other buffer --
        if (t + 1 < CRF_S) {
            const unsigned dst = ldsFeatBase + (unsigned)((buf ^ 1) * FEAT_BYTES) + tid * 32u;
            asm volatile(
                "global_load_async_to_lds_b128 %0, %1, off\n\t"
                "global_load_async_to_lds_b128 %0, %1, off offset:16"
                :: "v"(dst), "v"(gaddr) : "memory");
            gaddr += (unsigned long long)CRF_T * sizeof(float);
        }

        // -- per-row max (16 rows x 8 partials) -----------------------------
        {
            const int row = tid >> 3;
            const int p   = tid & 7;
            float mx = -INFINITY;
            #pragma unroll
            for (int c = 0; c < 8; ++c) mx = fmaxf(mx, sh_alpha[row][p * 8 + c]);
            sh_pmax[row][p] = mx;
        }
        __syncthreads();
        if (tid < ROWS) {
            float mx = sh_pmax[tid][0];
            #pragma unroll
            for (int p = 1; p < 8; ++p) mx = fmaxf(mx, sh_pmax[tid][p]);
            sh_rowmax[tid] = mx;
            sh_mrowf[tid]  = sh_mask[tid * CRF_S + t] ? 1.0f : 0.0f;
        }
        __syncthreads();

        // -- expA = exp(alpha - rowmax) as f16 ------------------------------
        #pragma unroll
        for (int e = 0; e < 8; ++e) {
            const int idx = tid * 8 + e;
            const int rr  = idx >> 6;
            const int cc  = idx & 63;
            sh_expA[rr][cc] = (_Float16)__expf(sh_alpha[rr][cc] - sh_rowmax[rr]);
        }
        // own f_t loads (issued last iter) are in-order older than the two
        // just-issued t+1 loads -> asynccnt<=2 means they are complete.
        if (t + 1 < CRF_S) asm volatile("s_wait_asynccnt 0x2" ::: "memory");
        else               asm volatile("s_wait_asynccnt 0x0" ::: "memory");
        __syncthreads();   // f_t tile + expA now block-visible

        // -- A fragments (K 0..31 and 32..63), two 16B LDS reads each -------
        v16h a0, a1;
        {
            const _Float16* rowp = &sh_expA[Arow][0];
            const int base = laneHi * 8;
            v8h l0 = *(const v8h*)(rowp + base);          // K = base..base+7
            v8h h0 = *(const v8h*)(rowp + base + 16);     // K = base+16..+23
            v8h l1 = *(const v8h*)(rowp + base + 32);
            v8h h1 = *(const v8h*)(rowp + base + 48);
            #pragma unroll
            for (int i = 0; i < 8; ++i) {
                a0[i] = l0[i]; a0[i + 8] = h0[i];
                a1[i] = l1[i]; a1[i + 8] = h1[i];
            }
        }

        // -- exp-space GEMM: acc = A0*B0 + A1*B1 ----------------------------
        v8f acc = {};
        acc = __builtin_amdgcn_wmma_f32_16x16x32_f16(false, a0, false, bf0,
                                                     (short)0, acc, false, false);
        acc = __builtin_amdgcn_wmma_f32_16x16x32_f16(false, a1, false, bf1,
                                                     (short)0, acc, false, false);

        // -- new alpha = rowmax + ln2*log2(acc) + f_t; branchless mask ------
        const float* fr = &sh_feat[buf][0][0];
        float newv[8];
        #pragma unroll
        for (int r = 0; r < 8; ++r) {
            const int M = r + laneHi * 8;
            const int N = Nbase + nn;
            const float v = sh_rowmax[M]
                          + LN2F * __builtin_amdgcn_logf(acc[r])   // acc >= ~e^-5
                          + fr[M * CRF_T + N];
            const float old = sh_alpha[M][N];
            newv[r] = old + sh_mrowf[M] * (v - old);
        }
        __syncthreads();   // everyone done reading old alpha / expA / f_t
        #pragma unroll
        for (int r = 0; r < 8; ++r) {
            const int M = r + laneHi * 8;
            sh_alpha[M][Nbase + nn] = newv[r];
        }
        __syncthreads();   // alpha' visible for next step
    }

    // ---- fwd = logsumexp(alpha + end); accumulate fwd - gold --------------
    if (tid < ROWS) {
        float mx = -INFINITY;
        #pragma unroll
        for (int n = 0; n < CRF_T; ++n) mx = fmaxf(mx, sh_alpha[tid][n] + endt[n]);
        float ssum = 0.0f;
        #pragma unroll
        for (int n = 0; n < CRF_T; ++n) ssum += __expf(sh_alpha[tid][n] + endt[n] - mx);
        const float fwd = mx + LN2F * __builtin_amdgcn_logf(ssum);
        atomicAdd(out, fwd - gold[bb + tid]);
    }
}

// ---------------------------------------------------------------------------
extern "C" void kernel_launch(void* const* d_in, const int* in_sizes, int n_in,
                              void* d_out, int out_size, void* d_ws, size_t ws_size,
                              hipStream_t stream) {
    const float*         feats  = (const float*)d_in[0];
    const float*         trans  = (const float*)d_in[1];
    const float*         startt = (const float*)d_in[2];
    const float*         endt   = (const float*)d_in[3];
    const int*           tags   = (const int*)d_in[4];
    const unsigned char* mask   = (const unsigned char*)d_in[5];
    float*               out    = (float*)d_out;
    float*               gold   = (float*)d_ws;          // B floats

    hipMemsetAsync(out, 0, sizeof(float), stream);

    crf_gold_kernel<<<CRF_B, 256, 0, stream>>>(feats, trans, startt, endt,
                                               tags, mask, gold);
    crf_forward_kernel<<<CRF_B / ROWS, 128, 0, stream>>>(feats, trans, startt,
                                                         endt, mask, gold, out);
}
